// ClassTransformer_60756607369371
// MI455X (gfx1250) — matrix operand order repaired and checked by reading, not compiled, not run on hardware
//
#include <hip/hip_runtime.h>
#include <hip/hip_bf16.h>

#define DM    256
#define DM2   512
#define NH    8
#define HDIM  32
#define NLAY  4
#define BATCH 8
#define SEQ   4096
#define KCLS  10
#define MTOK  (BATCH*SEQ)   // 32768

typedef __attribute__((ext_vector_type(16))) _Float16 v16h;
typedef __attribute__((ext_vector_type(8)))  _Float16 v8h;
typedef __attribute__((ext_vector_type(8)))  float    v8f;
typedef __attribute__((ext_vector_type(4)))  unsigned int u32x4;
typedef __attribute__((ext_vector_type(8)))  int      i32x8;
typedef __attribute__((ext_vector_type(4)))  int      i32x4;

union Frag16 { v16h v; v8h h[2]; };

// ---------------------------------------------------------------------------
// Cosine-similarity top-1 routing: seg[b,l] = argmax_c cos(fc[b,c], x[b,l])
// ---------------------------------------------------------------------------
__global__ __launch_bounds__(256) void route_kernel(
    const float* __restrict__ feat, const float* __restrict__ fcls,
    int* __restrict__ seg)
{
  const int b  = blockIdx.x >> 4;            // SEQ/256 = 16 blocks per batch
  const int l0 = (blockIdx.x & 15) * 256;
  const int t  = threadIdx.x;
  __shared__ float cls[DM][KCLS];            // [d][c] for broadcast reads
  __shared__ float na[KCLS];
  for (int i = t; i < KCLS * DM; i += 256) {
    int c = i >> 8, d = i & 255;
    cls[d][c] = fcls[(size_t)(b * KCLS + c) * DM + d];
  }
  __syncthreads();
  if (t < KCLS) {
    float s = 0.f;
    for (int d = 0; d < DM; d++) { float v = cls[d][t]; s += v * v; }
    na[t] = sqrtf(s);
  }
  __syncthreads();
  const int l = l0 + t;
  const float* xp = feat + (size_t)(b * SEQ + l) * DM;
  float dot[KCLS];
  #pragma unroll
  for (int c = 0; c < KCLS; c++) dot[c] = 0.f;
  float nb2 = 0.f;
  for (int d = 0; d < DM; d++) {
    float xv = xp[d];
    nb2 += xv * xv;
    #pragma unroll
    for (int c = 0; c < KCLS; c++) dot[c] += xv * cls[d][c];
  }
  float nb = sqrtf(nb2);
  int best = 0; float bests = -3.4e38f;
  #pragma unroll
  for (int c = 0; c < KCLS; c++) {
    float s = dot[c] / fmaxf(na[c] * nb, 1e-8f);
    if (s > bests) { bests = s; best = c; }    // first-max (jnp.argmax) semantics
  }
  seg[b * SEQ + l] = best;
}

// ---------------------------------------------------------------------------
// x init: fp32 copy + f16 shadow for WMMA A-operand
// ---------------------------------------------------------------------------
__global__ __launch_bounds__(256) void init_x(
    const float* __restrict__ feat, float* __restrict__ xf32,
    _Float16* __restrict__ xf16)
{
  size_t i = (size_t)blockIdx.x * 256 + threadIdx.x;
  float v = feat[i];
  xf32[i] = v;
  xf16[i] = (_Float16)v;
}

// Convert + transpose weights: src f32 [Kd,N] row-major -> dst f16 [N,Kd]
__global__ __launch_bounds__(256) void cvtT(
    const float* __restrict__ src, _Float16* __restrict__ dst, int Kd, int N)
{
  size_t i = (size_t)blockIdx.x * 256 + threadIdx.x;   // over Kd*N
  int k = (int)(i / N), n = (int)(i % N);
  dst[(size_t)n * Kd + k] = (_Float16)src[i];
}

// ---------------------------------------------------------------------------
// WMMA GEMM: C[M,N] = A[M,Kd] * Bt[N,Kd]^T   (A row-major f16, B pre-transposed)
// The block's 64xKd weight slab is DMA'd into LDS once by the Tensor Data
// Mover (tensor_load_to_lds, TENSORcnt), then all 8 waves feed their B
// fragments from LDS (ds_load_b128) while A streams from global with
// prefetch.  8 waves/block, wave tile 32x32 (2x2 v_wmma), block 128x64.
// ACT: 0=none, 1=relu, 2=elu+1.  OUTH: store f16 (else f32).
// ---------------------------------------------------------------------------
template<int ACT, bool OUTH>
__device__ inline void store_tile(const v8f& c, int mt, int nt,
                                  float* Cf, _Float16* Ch, int N, int lane)
{
  int n  = nt + (lane & 15);
  int mr = mt + (lane >> 4) * 8;               // C/D layout: vgpr i -> row mr+i
  #pragma unroll
  for (int i = 0; i < 8; i++) {
    float x = c[i];
    if (ACT == 1) x = fmaxf(x, 0.f);
    else if (ACT == 2) x = (x > 0.f) ? (x + 1.f) : __expf(x);  // elu(x)+1
    if (OUTH) Ch[(size_t)(mr + i) * N + n] = (_Float16)x;
    else      Cf[(size_t)(mr + i) * N + n] = x;
  }
}

template<int ACT, bool OUTH>
__global__ __launch_bounds__(256) void gemm_wmma(
    const _Float16* __restrict__ A, const _Float16* __restrict__ Bt,
    float* __restrict__ Cf, _Float16* __restrict__ Ch, int N, int Kd)
{
  __shared__ __align__(16) _Float16 bsh[64 * DM2];   // 64 cols x Kd<=512: 64 KB
  const int lane = threadIdx.x & 31;
  const int wave = threadIdx.x >> 5;
  const int m0   = blockIdx.x * 128 + (wave & 3) * 32;
  const int nblk = blockIdx.y * 64;                  // global col base of block
  const int n0l  = (wave >> 2) * 32;                 // wave col base inside LDS
  const int r15  = lane & 15;
  const int half = lane >> 4;

  // --- TDM: DMA Bt[nblk .. nblk+63][0..Kd) into LDS (one wave issues) ------
  if (wave == 0) {
    unsigned long long ga = (unsigned long long)(uintptr_t)(Bt + (size_t)nblk * Kd);
    unsigned int la = (unsigned int)(uintptr_t)bsh;  // flat low 32b == LDS byte addr
    u32x4 g0;
    g0[0] = 1u;                                  // count=1, no gather, user D#
    g0[1] = la;                                  // lds_addr
    g0[2] = (unsigned int)ga;                    // global_addr[31:0]
    g0[3] = (unsigned int)(ga >> 32) | 0x80000000u;  // global_addr[56:32] | type=2
    i32x8 g1;
    g1[0] = 1 << 16;                             // data_size=1 (2 bytes)
    g1[1] = (Kd & 0xFFFF) << 16;                 // tensor_dim0[15:0] (bits 63:48)
    g1[2] = ((unsigned)Kd >> 16) | ((N & 0xFFFF) << 16);   // dim0 hi / dim1 lo
    g1[3] = ((unsigned)N >> 16) | ((Kd & 0xFFFF) << 16);   // dim1 hi / tile_dim0=Kd
    g1[4] = 64;                                  // tile_dim1=64, tile_dim2=0
    g1[5] = Kd;                                  // tensor_dim0_stride[31:0]
    g1[6] = 0;                                   // stride0 hi / stride1 lo
    g1[7] = 0;                                   // stride1 hi
    i32x4 g2 = {0, 0, 0, 0};                     // 2-D tile: groups 2/3 unused
    i32x4 g3 = {0, 0, 0, 0};
    i32x8 g4 = {0, 0, 0, 0, 0, 0, 0, 0};         // 6-arg toolchain: extra group
    __builtin_amdgcn_tensor_load_to_lds(g0, g1, g2, g3, g4, 0);
    __builtin_amdgcn_s_wait_tensorcnt(0);
  }
  __syncthreads();

  // A fragment (16x32 f16): lane row = lane&15, kbase = 8*half,
  //   halves j=0..7 -> K=kbase+j ; j=8..15 -> K=kbase+16+(j-8)
  const _Float16* a0p = A  + (size_t)(m0 + r15) * Kd + half * 8;
  const _Float16* a1p = a0p + (size_t)16 * Kd;
  // B fragment (32x16 f16) from the LDS slab: lane col = lane&15,
//   kbase = 8*half, halves j=0..7 -> K=kbase+j ; j=8..15 -> K=kbase+16+(j-8)
  const _Float16* b0p = bsh + (size_t)(n0l + r15) * Kd + half * 8;
  const _Float16* b1p = b0p + (size_t)16 * Kd;

  v8f acc00 = {}, acc01 = {}, acc10 = {}, acc11 = {};

  for (int kq = 0; kq < Kd; kq += 32) {
    Frag16 a0, a1, b0, b1;
    a0.h[0] = *(const v8h*)(a0p + kq);
    a0.h[1] = *(const v8h*)(a0p + kq + 16);
    a1.h[0] = *(const v8h*)(a1p + kq);
    a1.h[1] = *(const v8h*)(a1p + kq + 16);
    b0.h[0] = *(const v8h*)(b0p + kq);           // ds_load_b128
    b0.h[1] = *(const v8h*)(b0p + kq + 16);
    b1.h[0] = *(const v8h*)(b1p + kq);
    b1.h[1] = *(const v8h*)(b1p + kq + 16);
    if (kq + 32 < Kd) {                          // global_prefetch_b8 next A slab
      __builtin_prefetch(a0p + kq + 32, 0, 1);
      __builtin_prefetch(a1p + kq + 32, 0, 1);
    }
    acc00 = __builtin_amdgcn_wmma_f32_16x16x32_f16(false, a0.v, false, b0.v,
                                                   (short)0, acc00, false, false);
    acc01 = __builtin_amdgcn_wmma_f32_16x16x32_f16(false, a0.v, false, b1.v,
                                                   (short)0, acc01, false, false);
    acc10 = __builtin_amdgcn_wmma_f32_16x16x32_f16(false, a1.v, false, b0.v,
                                                   (short)0, acc10, false, false);
    acc11 = __builtin_amdgcn_wmma_f32_16x16x32_f16(false, a1.v, false, b1.v,
                                                   (short)0, acc11, false, false);
  }
  const int nw = nblk + n0l;
  store_tile<ACT, OUTH>(acc00, m0,      nw,      Cf, Ch, N, lane);
  store_tile<ACT, OUTH>(acc01, m0,      nw + 16, Cf, Ch, N, lane);
  store_tile<ACT, OUTH>(acc10, m0 + 16, nw,      Cf, Ch, N, lane);
  store_tile<ACT, OUTH>(acc11, m0 + 16, nw + 16, Cf, Ch, N, lane);
}

// ---------------------------------------------------------------------------
// Per-(batch,head) segment reduction in LDS:
//   KV[b,c,h,d,e] = sum_{l: seg=c} Kf[l,d]*v[l,e];  Ksum[b,c,h,d] = sum Kf[l,d]
// Exclusive (d,e) ownership per thread -> plain LDS RMW, no atomics.
// ---------------------------------------------------------------------------
__global__ __launch_bounds__(256) void seg_reduce(
    const float* __restrict__ kf, const float* __restrict__ vv,
    const int* __restrict__ seg, float* __restrict__ KV,
    float* __restrict__ Ksum)
{
  const int b = blockIdx.x >> 3;
  const int h = blockIdx.x & (NH - 1);
  const int t = threadIdx.x;
  __shared__ float kv_s[KCLS][HDIM][HDIM];   // 40 KB
  __shared__ float ks_s[KCLS][HDIM];
  __shared__ float kf_s[64][HDIM];
  __shared__ float v_s[64][HDIM];
  __shared__ int   seg_s[64];
  for (int i = t; i < KCLS * HDIM * HDIM; i += 256) ((float*)kv_s)[i] = 0.f;
  for (int i = t; i < KCLS * HDIM; i += 256) ((float*)ks_s)[i] = 0.f;
  const int e = t & 31;
  const int dbase = t >> 5;                  // 0..7
  for (int l0 = 0; l0 < SEQ; l0 += 64) {
    __syncthreads();
    #pragma unroll
    for (int j = 0; j < 8; j++) {
      int lin = t + 256 * j; int tok = lin >> 5; int d = lin & 31;
      size_t g = (size_t)(b * SEQ + l0 + tok) * DM + h * HDIM + d;
      kf_s[tok][d] = kf[g];
      v_s[tok][d]  = vv[g];
    }
    if (t < 64) seg_s[t] = seg[b * SEQ + l0 + t];
    __syncthreads();
    for (int tok = 0; tok < 64; tok++) {
      int c = seg_s[tok];
      float ve = v_s[tok][e];
      #pragma unroll
      for (int i = 0; i < 4; i++) {
        int d = dbase + 8 * i;                // thread owns (d,e) exclusively
        kv_s[c][d][e] += kf_s[tok][d] * ve;
      }
      if (t < 32) ks_s[c][e] += kf_s[tok][e]; // lane owns d=e exclusively
    }
  }
  __syncthreads();
  for (int i = t; i < KCLS * HDIM * HDIM; i += 256) {
    int c = i >> 10; int de = i & 1023;
    KV[(size_t)((b * KCLS + c) * NH + h) * (HDIM * HDIM) + de] = ((float*)kv_s)[i];
  }
  for (int i = t; i < KCLS * HDIM; i += 256) {
    int c = i >> 5; int d = i & 31;
    Ksum[(size_t)((b * KCLS + c) * NH + h) * HDIM + d] = ks_s[c][d];
  }
}

// ---------------------------------------------------------------------------
// msg[b,l,h,:] = (Q . KV[seg]) / (Q . Ksum[seg] + eps), emitted as f16
// ---------------------------------------------------------------------------
__global__ __launch_bounds__(256) void apply_attn(
    const float* __restrict__ q, const float* __restrict__ KV,
    const float* __restrict__ Ksum, const int* __restrict__ seg,
    _Float16* __restrict__ msg)
{
  int idx  = blockIdx.x * 256 + threadIdx.x;  // over MTOK*NH
  int h    = idx & (NH - 1);
  int tokn = idx >> 3;                        // b*SEQ + l
  int b    = tokn >> 12;
  int c    = seg[tokn];
  const float* qp  = q + (size_t)tokn * DM + h * HDIM;
  const float* kvp = KV + (size_t)((b * KCLS + c) * NH + h) * (HDIM * HDIM);
  const float* ksp = Ksum + (size_t)((b * KCLS + c) * NH + h) * HDIM;
  float acc[HDIM];
  #pragma unroll
  for (int e = 0; e < HDIM; e++) acc[e] = 0.f;
  float zden = 0.f;
  for (int d = 0; d < HDIM; d++) {
    float qd = qp[d];
    zden += qd * ksp[d];
    #pragma unroll
    for (int e = 0; e < HDIM; e++) acc[e] += qd * kvp[d * HDIM + e];
  }
  float z = 1.f / (zden + 1e-6f);
  _Float16* mp = msg + (size_t)tokn * DM + h * HDIM;
  #pragma unroll
  for (int e = 0; e < HDIM; e++) mp[e] = (_Float16)(acc[e] * z);
}

// ---------------------------------------------------------------------------
// Wave-per-row layernorms (wave32 shuffle reductions)
// ---------------------------------------------------------------------------
__device__ inline float wave_sum32(float x) {
  #pragma unroll
  for (int off = 16; off > 0; off >>= 1) x += __shfl_xor(x, off, 32);
  return x;
}

// LN(m; g1,b1) then build xm = [x | LN(m)] in f16 for the FFN GEMM
__global__ __launch_bounds__(256) void ln_concat(
    const float* __restrict__ m, const float* __restrict__ g,
    const float* __restrict__ bb, const float* __restrict__ x,
    _Float16* __restrict__ xm)
{
  int wave = threadIdx.x >> 5, lane = threadIdx.x & 31;
  int row  = blockIdx.x * 8 + wave;
  const float* mp = m + (size_t)row * DM;
  float vals[8]; float s = 0.f;
  #pragma unroll
  for (int j = 0; j < 8; j++) { vals[j] = mp[lane + 32 * j]; s += vals[j]; }
  float mu = wave_sum32(s) * (1.f / DM);
  float s2 = 0.f;
  #pragma unroll
  for (int j = 0; j < 8; j++) { vals[j] -= mu; s2 += vals[j] * vals[j]; }
  float rstd = rsqrtf(wave_sum32(s2) * (1.f / DM) + 1e-5f);
  _Float16* xp = xm + (size_t)row * DM2;
  const float* xr = x + (size_t)row * DM;
  #pragma unroll
  for (int j = 0; j < 8; j++) {
    int d = lane + 32 * j;
    xp[d]      = (_Float16)xr[d];
    xp[DM + d] = (_Float16)(g[d] * vals[j] * rstd + bb[d]);
  }
}

// x += LN(h2; g2,b2); refresh f16 shadow of x
__global__ __launch_bounds__(256) void ln_residual(
    const float* __restrict__ h2, const float* __restrict__ g,
    const float* __restrict__ bb, float* __restrict__ x,
    _Float16* __restrict__ x16)
{
  int wave = threadIdx.x >> 5, lane = threadIdx.x & 31;
  int row  = blockIdx.x * 8 + wave;
  const float* hp = h2 + (size_t)row * DM;
  float vals[8]; float s = 0.f;
  #pragma unroll
  for (int j = 0; j < 8; j++) { vals[j] = hp[lane + 32 * j]; s += vals[j]; }
  float mu = wave_sum32(s) * (1.f / DM);
  float s2 = 0.f;
  #pragma unroll
  for (int j = 0; j < 8; j++) { vals[j] -= mu; s2 += vals[j] * vals[j]; }
  float rstd = rsqrtf(wave_sum32(s2) * (1.f / DM) + 1e-5f);
  float* xr = x + (size_t)row * DM;
  _Float16* xs = x16 + (size_t)row * DM;
  #pragma unroll
  for (int j = 0; j < 8; j++) {
    int d = lane + 32 * j;
    float o = xr[d] + (g[d] * vals[j] * rstd + bb[d]);
    xr[d] = o;
    xs[d] = (_Float16)o;
  }
}

__global__ __launch_bounds__(256) void copy_out(
    const float* __restrict__ x, float* __restrict__ out)
{
  size_t i = (size_t)blockIdx.x * 256 + threadIdx.x;
  out[i] = x[i];
}

// ---------------------------------------------------------------------------
extern "C" void kernel_launch(void* const* d_in, const int* in_sizes, int n_in,
                              void* d_out, int out_size, void* d_ws, size_t ws_size,
                              hipStream_t stream) {
  const float* feat = (const float*)d_in[1];
  const float* fcls = (const float*)d_in[2];
  const float* Wq = (const float*)d_in[3];
  const float* Wk = (const float*)d_in[4];
  const float* Wv = (const float*)d_in[5];
  const float* Wm = (const float*)d_in[6];
  const float* W1 = (const float*)d_in[7];
  const float* W2 = (const float*)d_in[8];
  const float* g1 = (const float*)d_in[9];
  const float* b1 = (const float*)d_in[10];
  const float* g2 = (const float*)d_in[11];
  const float* b2 = (const float*)d_in[12];

  char* w = (char*)d_ws;
  auto alloc = [&](size_t bytes) -> char* {
    char* p = w; w += (bytes + 255) & ~(size_t)255; return p;
  };
  int*      seg   = (int*)alloc((size_t)MTOK * sizeof(int));
  float*    xf32  = (float*)alloc((size_t)MTOK * DM * 4);
  _Float16* xf16  = (_Float16*)alloc((size_t)MTOK * DM * 2);
  float*    qf32  = (float*)alloc((size_t)MTOK * DM * 4);
  float*    kf32  = (float*)alloc((size_t)MTOK * DM * 4);
  float*    vf32  = (float*)alloc((size_t)MTOK * DM * 4);
  _Float16* msg16 = (_Float16*)alloc((size_t)MTOK * DM * 2);
  _Float16* h116  = (_Float16*)alloc((size_t)MTOK * DM2 * 2);
  float*    KV    = (float*)alloc((size_t)BATCH * KCLS * NH * HDIM * HDIM * 4);
  float*    Ksum  = (float*)alloc((size_t)BATCH * KCLS * NH * HDIM * 4);
  _Float16* wq16  = (_Float16*)alloc((size_t)DM * DM * 2);
  _Float16* wk16  = (_Float16*)alloc((size_t)DM * DM * 2);
  _Float16* wv16  = (_Float16*)alloc((size_t)DM * DM * 2);
  _Float16* wm16  = (_Float16*)alloc((size_t)DM * DM * 2);
  _Float16* w116  = (_Float16*)alloc((size_t)DM2 * DM2 * 2);
  _Float16* w216  = (_Float16*)alloc((size_t)DM2 * DM * 2);
  // lifetime-safe aliases (k dead after seg_reduce, v after seg_reduce,
  // q after apply_attn)
  float*    mf32  = kf32;
  float*    h2f32 = vf32;
  _Float16* xm16  = (_Float16*)qf32;        // MTOK*512*2 == MTOK*256*4 bytes

  route_kernel<<<BATCH * (SEQ / 256), 256, 0, stream>>>(feat, fcls, seg);
  init_x<<<(MTOK * DM) / 256, 256, 0, stream>>>(feat, xf32, xf16);

  dim3 gN4(MTOK / 128, DM / 64);            // N=256 GEMMs
  dim3 gN8(MTOK / 128, DM2 / 64);           // N=512 GEMM
  for (int li = 0; li < NLAY; li++) {
    cvtT<<<(DM * DM) / 256, 256, 0, stream>>>(Wq + (size_t)li * DM * DM, wq16, DM, DM);
    cvtT<<<(DM * DM) / 256, 256, 0, stream>>>(Wk + (size_t)li * DM * DM, wk16, DM, DM);
    cvtT<<<(DM * DM) / 256, 256, 0, stream>>>(Wv + (size_t)li * DM * DM, wv16, DM, DM);
    cvtT<<<(DM * DM) / 256, 256, 0, stream>>>(Wm + (size_t)li * DM * DM, wm16, DM, DM);
    cvtT<<<(DM2 * DM2) / 256, 256, 0, stream>>>(W1 + (size_t)li * DM2 * DM2, w116, DM2, DM2);
    cvtT<<<(DM2 * DM) / 256, 256, 0, stream>>>(W2 + (size_t)li * DM2 * DM, w216, DM2, DM);

    // Q/K with fused elu+1 epilogue, V plain
    gemm_wmma<2, false><<<gN4, 256, 0, stream>>>(xf16, wq16, qf32, nullptr, DM, DM);
    gemm_wmma<2, false><<<gN4, 256, 0, stream>>>(xf16, wk16, kf32, nullptr, DM, DM);
    gemm_wmma<0, false><<<gN4, 256, 0, stream>>>(xf16, wv16, vf32, nullptr, DM, DM);

    seg_reduce<<<BATCH * NH, 256, 0, stream>>>(kf32, vf32, seg, KV, Ksum);
    apply_attn<<<(MTOK * NH) / 256, 256, 0, stream>>>(qf32, KV, Ksum, seg, msg16);

    gemm_wmma<0, false><<<gN4, 256, 0, stream>>>(msg16, wm16, mf32, nullptr, DM, DM);
    ln_concat<<<MTOK / 8, 256, 0, stream>>>(mf32, g1 + li * DM, b1 + li * DM, xf32, xm16);
    gemm_wmma<1, true><<<gN8, 256, 0, stream>>>(xm16, w116, nullptr, h116, DM2, DM2);
    gemm_wmma<0, false><<<gN4, 256, 0, stream>>>(h116, w216, h2f32, nullptr, DM, DM2);
    ln_residual<<<MTOK / 8, 256, 0, stream>>>(h2f32, g2 + li * DM, b2 + li * DM, xf32, xf16);
  }
  copy_out<<<(MTOK * DM) / 256, 256, 0, stream>>>(xf32, (float*)d_out);
}
